// MPSCompatibleTransformerLayer_317827580307
// MI455X (gfx1250) — matrix operand order repaired and checked by reading, not compiled, not run on hardware
//
#include <hip/hip_runtime.h>
#include <hip/hip_bf16.h>
#include <math.h>

// ---------------------------------------------------------------------------
// MI455X / gfx1250 transformer layer: bf16 WMMA GEMMs + flash attention,
// with GLOBAL_LOAD_ASYNC_TO_LDS_B128 double-buffered tile staging (ASYNCcnt).
// Wave32, WMMA 16x16x32 bf16 -> f32 accumulate.
// ---------------------------------------------------------------------------

#define HDIM   1024
#define NHEAD  16
#define HD     64
#define FFDIM  4096
#define BB     4
#define SS     2048
#define BS     (BB * SS)          // 8192 rows

typedef __bf16 bf16;
typedef __attribute__((ext_vector_type(16))) __bf16 v16bf;
typedef __attribute__((ext_vector_type(8)))  __bf16 v8bf;
typedef __attribute__((ext_vector_type(8)))  float  v8f;

__device__ __forceinline__ v8f wmma_bf16(v16bf a, v16bf b, v8f c) {
  // v_wmma_f32_16x16x32_bf16  (neg_a, A, neg_b, B, c_mod, C, reuse_a, reuse_b)
  return __builtin_amdgcn_wmma_f32_16x16x32_bf16(false, a, false, b, (short)0, c,
                                                 false, false);
}

__device__ __forceinline__ v16bf cat8(v8bf lo, v8bf hi) {
  return __builtin_shufflevector(lo, hi, 0, 1, 2, 3, 4, 5, 6, 7,
                                 8, 9, 10, 11, 12, 13, 14, 15);
}

// LDS byte address = low 32 bits of the generic pointer
__device__ __forceinline__ unsigned lds_addr(const void* p) {
  return (unsigned)(uintptr_t)p;
}

// Async DMA: memory -> LDS, 16 bytes per lane, tracked by ASYNCcnt.
__device__ __forceinline__ void async_b128(unsigned lds, const void* gaddr) {
  asm volatile("global_load_async_to_lds_b128 %0, %1, off"
               :: "v"(lds), "v"(gaddr)
               : "memory");
}
__device__ __forceinline__ void wait_async0() {
  asm volatile("s_wait_asynccnt 0" ::: "memory");
}

// ---------------------------------------------------------------------------
// f32 -> bf16 elementwise convert
// ---------------------------------------------------------------------------
__global__ __launch_bounds__(256) void k_to_bf16(const float* __restrict__ in,
                                                 bf16* __restrict__ out,
                                                 size_t n) {
  size_t i = (size_t)blockIdx.x * 256 + threadIdx.x;
  if (i < n) out[i] = (bf16)in[i];
}

// ---------------------------------------------------------------------------
// f32 [R,C] -> bf16 transposed [C,R]  (weights into [N,K] for GEMM B operand)
// ---------------------------------------------------------------------------
__global__ __launch_bounds__(256) void k_transpose_bf16(
    const float* __restrict__ in, bf16* __restrict__ out, int R, int C) {
  __shared__ bf16 tile[32][33];
  int bc = blockIdx.x * 32;                 // column tile in 'in'
  int br = blockIdx.y * 32;                 // row tile in 'in'
  int tx = threadIdx.x & 31;
  int ty = threadIdx.x >> 5;                // 0..7
#pragma unroll
  for (int yy = ty; yy < 32; yy += 8)
    tile[yy][tx] = (bf16)in[(size_t)(br + yy) * C + (bc + tx)];
  __syncthreads();
#pragma unroll
  for (int yy = ty; yy < 32; yy += 8)
    out[(size_t)(bc + yy) * R + (br + tx)] = tile[tx][yy];
}

// ---------------------------------------------------------------------------
// GEMM: out[M,N] = A[M,K](bf16) @ BT[N,K](bf16)^T + bias, optional exact GELU.
// Block = 256 threads = 8 waves; tile 128x128, k-step 64, double-buffered LDS
// filled by global_load_async_to_lds_b128 (overlaps with WMMA of current tile).
// Wave (wr in 0..1, wc in 0..3) owns a 64x32 sub-tile = 4x2 WMMA tiles.
// ---------------------------------------------------------------------------
template <bool GELU>
__global__ __launch_bounds__(256) void k_gemm_bf16(
    const bf16* __restrict__ A, const bf16* __restrict__ BT,
    const float* __restrict__ bias, float* __restrict__ outF,
    bf16* __restrict__ outB, int M, int N, int K) {
  __shared__ bf16 As[2][128 * 64];
  __shared__ bf16 Bs[2][128 * 64];   // N-major: Bs[n][k]

  const int tid   = threadIdx.x;
  const int wave  = tid >> 5;
  const int lane  = tid & 31;
  const int l16   = lane & 15;
  const int lhalf = lane >> 4;
  const int wr    = wave >> 2;    // 0..1
  const int wc    = wave & 3;     // 0..3
  const int m0    = blockIdx.y * 128;
  const int n0    = blockIdx.x * 128;

  v8f acc[4][2];
#pragma unroll
  for (int i = 0; i < 4; ++i)
#pragma unroll
    for (int j = 0; j < 2; ++j) acc[i][j] = (v8f){};

  // Tile stage: 128x64 bf16 = 1024 x 16B chunks; each thread issues 4 per matrix.
  auto issue = [&](int buf, int k0) {
#pragma unroll
    for (int c = 0; c < 4; ++c) {
      int id  = tid + 256 * c;      // 0..1023
      int row = id >> 3;            // 0..127
      int col = (id & 7) * 8;       // 0..56 (bf16 elems)
      async_b128(lds_addr(&As[buf][row * 64 + col]),
                 &A[(size_t)(m0 + row) * K + k0 + col]);
      async_b128(lds_addr(&Bs[buf][row * 64 + col]),
                 &BT[(size_t)(n0 + row) * K + k0 + col]);
    }
  };

  const int nst = K >> 6;
  issue(0, 0);
  for (int s = 0; s < nst; ++s) {
    const int cur = s & 1;
    wait_async0();        // my loads for buffer 'cur' have landed in LDS
    __syncthreads();      // everyone's loads landed; everyone done reading cur^1
    if (s + 1 < nst) issue(cur ^ 1, (s + 1) * 64);  // overlap next fill

    const bf16* as = &As[cur][0];
    const bf16* bs = &Bs[cur][0];
#pragma unroll
    for (int half = 0; half < 2; ++half) {
      // B fragments: lane holds column n, K contiguous (2 per VGPR).
      v16bf bfrag[2];
#pragma unroll
      for (int j = 0; j < 2; ++j) {
        int n = wc * 32 + j * 16 + l16;
        bfrag[j] = *(const v16bf*)&bs[n * 64 + half * 32 + lhalf * 16];
      }
#pragma unroll
      for (int i = 0; i < 4; ++i) {
        // A fragment: lane row m; K contiguous [16*lhalf..+15].
        int m = wr * 64 + i * 16 + l16;
        v8bf lo = *(const v8bf*)&as[m * 64 + half * 32 + lhalf * 16];
        v8bf hi = *(const v8bf*)&as[m * 64 + half * 32 + lhalf * 16 + 8];
        v16bf afrag = cat8(lo, hi);
        acc[i][0] = wmma_bf16(afrag, bfrag[0], acc[i][0]);
        acc[i][1] = wmma_bf16(afrag, bfrag[1], acc[i][1]);
      }
    }
  }

  // Epilogue: C layout -> lane col n = l16 (+tile), row = r + 8*lhalf.
#pragma unroll
  for (int i = 0; i < 4; ++i)
#pragma unroll
    for (int j = 0; j < 2; ++j) {
      int n = n0 + wc * 32 + j * 16 + l16;
      float bv = bias ? bias[n] : 0.0f;
#pragma unroll
      for (int r = 0; r < 8; ++r) {
        int m = m0 + wr * 64 + i * 16 + r + 8 * lhalf;
        float v = acc[i][j][r] + bv;
        if (GELU) v = 0.5f * v * (1.0f + erff(v * 0.70710678118654752f));
        size_t idx = (size_t)m * N + n;
        if (outF) outF[idx] = v;
        if (outB) outB[idx] = (bf16)v;
      }
    }
}

// ---------------------------------------------------------------------------
// Flash attention: block = 256 threads = 8 waves; block owns (b,h) and 128
// query rows (16 per wave). Loop over key blocks of 32; K-tile staged with
// async-to-LDS; V staged d-major via VALU transpose; P transposed via
// per-wave LDS scratch (DS ops are in-order within a wave).
// ---------------------------------------------------------------------------
__global__ __launch_bounds__(256) void k_attention(
    const bf16* __restrict__ qb, const bf16* __restrict__ kb,
    const bf16* __restrict__ vb, const float* __restrict__ mask,
    bf16* __restrict__ cb) {
  const int idx  = blockIdx.x;
  const int qblk = idx & (SS / 128 - 1);            // 0..15
  const int h    = (idx >> 4) & (NHEAD - 1);        // 0..15
  const int b    = idx >> 8;                        // 0..3
  const int tid = threadIdx.x, wave = tid >> 5, lane = tid & 31;
  const int l16 = lane & 15, lhalf = lane >> 4;
  const int q0 = qblk * 128 + wave * 16;

  __shared__ bf16 Ks[32 * 64];        // [key][d]
  __shared__ bf16 Vt[64 * 32];        // [d][key]  (transposed)
  __shared__ bf16 Ps[8][16 * 32];     // per-wave P scratch [m][key]

  // Q fragments for this wave's 16 rows (held for the whole loop).
  const bf16* qptr = qb + ((size_t)(b * SS + q0)) * HDIM + h * HD;
  v16bf qf[2];
#pragma unroll
  for (int s = 0; s < 2; ++s) {
    const bf16* p = qptr + (size_t)l16 * HDIM + s * 32 + lhalf * 16;
    qf[s] = cat8(*(const v8bf*)p, *(const v8bf*)(p + 8));
  }

  v8f acc[4];
#pragma unroll
  for (int t = 0; t < 4; ++t) acc[t] = (v8f){};
  float mrow[8], lrow[8];
#pragma unroll
  for (int r = 0; r < 8; ++r) { mrow[r] = -1.0e30f; lrow[r] = 0.0f; }

  const int krow = tid >> 3;            // 0..31 (key)
  const int kcol = (tid & 7) * 8;       // 0..56 (d)

  for (int kb0 = 0; kb0 < SS; kb0 += 32) {
    __syncthreads();   // previous tile fully consumed by all waves
    {  // stage K via async DMA; stage V d-major via VALU transpose
      size_t gro = ((size_t)(b * SS + kb0 + krow)) * HDIM + h * HD + kcol;
      async_b128(lds_addr(&Ks[krow * 64 + kcol]), &kb[gro]);
      v8bf vv = *(const v8bf*)&vb[gro];
#pragma unroll
      for (int e = 0; e < 8; ++e) Vt[(kcol + e) * 32 + krow] = vv[e];
    }
    wait_async0();
    __syncthreads();

    // scores S = Q @ K^T for two 16-key tiles
    v8f st[2];
#pragma unroll
    for (int j = 0; j < 2; ++j) {
      st[j] = (v8f){};
#pragma unroll
      for (int s = 0; s < 2; ++s) {
        v16bf kf =
            *(const v16bf*)&Ks[(j * 16 + l16) * 64 + s * 32 + lhalf * 16];
        st[j] = wmma_bf16(qf[s], kf, st[j]);
      }
    }
    // scale + additive mask (per key column)
    float madd[2];
#pragma unroll
    for (int j = 0; j < 2; ++j)
      madd[j] = (1.0f - mask[(size_t)b * SS + kb0 + j * 16 + l16]) * -10000.0f;
#pragma unroll
    for (int j = 0; j < 2; ++j)
#pragma unroll
      for (int r = 0; r < 8; ++r) st[j][r] = st[j][r] * 0.125f + madd[j];

    // online softmax row stats (rows live in VGPR index; reduce across the
    // 16 lanes of each half-wave)
    float tmax[8];
#pragma unroll
    for (int r = 0; r < 8; ++r) tmax[r] = fmaxf(st[0][r], st[1][r]);
#pragma unroll
    for (int d = 1; d < 16; d <<= 1)
#pragma unroll
      for (int r = 0; r < 8; ++r)
        tmax[r] = fmaxf(tmax[r], __shfl_xor(tmax[r], d, 32));

    float alpha[8], psum[8];
#pragma unroll
    for (int r = 0; r < 8; ++r) {
      float mn = fmaxf(mrow[r], tmax[r]);
      alpha[r] = __expf(mrow[r] - mn);
      mrow[r]  = mn;
      float p0 = __expf(st[0][r] - mn);
      float p1 = __expf(st[1][r] - mn);
      st[0][r] = p0; st[1][r] = p1;
      psum[r]  = p0 + p1;
    }
#pragma unroll
    for (int d = 1; d < 16; d <<= 1)
#pragma unroll
      for (int r = 0; r < 8; ++r) psum[r] += __shfl_xor(psum[r], d, 32);
#pragma unroll
    for (int r = 0; r < 8; ++r) lrow[r] = lrow[r] * alpha[r] + psum[r];
#pragma unroll
    for (int t = 0; t < 4; ++t)
#pragma unroll
      for (int r = 0; r < 8; ++r) acc[t][r] *= alpha[r];

    // transpose P (C layout) -> A-fragment layout via per-wave LDS
    bf16* ps = &Ps[wave][0];
#pragma unroll
    for (int j = 0; j < 2; ++j)
#pragma unroll
      for (int r = 0; r < 8; ++r)
        ps[(r + 8 * lhalf) * 32 + j * 16 + l16] = (bf16)st[j][r];
    asm volatile("" ::: "memory");  // DS ops are in-order within a wave

    v16bf pf;
    {
      v8bf lo = *(const v8bf*)&ps[l16 * 32 + lhalf * 16];
      v8bf hi = *(const v8bf*)&ps[l16 * 32 + lhalf * 16 + 8];
      pf = cat8(lo, hi);
    }
#pragma unroll
    for (int t = 0; t < 4; ++t) {
      v16bf vf = *(const v16bf*)&Vt[(t * 16 + l16) * 32 + lhalf * 16];
      acc[t] = wmma_bf16(pf, vf, acc[t]);
    }
  }

  // normalize and write ctx (bf16, [b, s, h*64 + d] layout)
  float inv[8];
#pragma unroll
  for (int r = 0; r < 8; ++r) inv[r] = 1.0f / lrow[r];
  bf16* out = cb + ((size_t)(b * SS + q0)) * HDIM + h * HD;
#pragma unroll
  for (int t = 0; t < 4; ++t)
#pragma unroll
    for (int r = 0; r < 8; ++r)
      out[(size_t)(r + 8 * lhalf) * HDIM + t * 16 + l16] =
          (bf16)(acc[t][r] * inv[r]);
}

// ---------------------------------------------------------------------------
// Residual + LayerNorm over H=1024. One block (256 thr) per row.
// ---------------------------------------------------------------------------
__global__ __launch_bounds__(256) void k_ln_residual(
    const float* __restrict__ x, const float* __restrict__ y,
    const float* __restrict__ g, const float* __restrict__ be,
    float* __restrict__ outF, bf16* __restrict__ outB) {
  const size_t row = blockIdx.x;
  const float* xr = x + row * HDIM;
  const float* yr = y + row * HDIM;
  const int tid = threadIdx.x, wave = tid >> 5, lane = tid & 31;
  float v[4], s = 0.0f, ss = 0.0f;
#pragma unroll
  for (int i = 0; i < 4; ++i) {
    int c = tid + i * 256;
    float t = xr[c] + yr[c];
    v[i] = t; s += t; ss += t * t;
  }
#pragma unroll
  for (int d = 1; d < 32; d <<= 1) {
    s += __shfl_xor(s, d, 32);
    ss += __shfl_xor(ss, d, 32);
  }
  __shared__ float red[16];
  if (lane == 0) { red[wave] = s; red[8 + wave] = ss; }
  __syncthreads();
  float stot = 0.0f, sstot = 0.0f;
#pragma unroll
  for (int i = 0; i < 8; ++i) { stot += red[i]; sstot += red[8 + i]; }
  float mu = stot * (1.0f / HDIM);
  float var = sstot * (1.0f / HDIM) - mu * mu;
  float rstd = rsqrtf(var + 1e-5f);
#pragma unroll
  for (int i = 0; i < 4; ++i) {
    int c = tid + i * 256;
    float o = (v[i] - mu) * rstd * g[c] + be[c];
    size_t idx = row * HDIM + c;
    if (outF) outF[idx] = o;
    if (outB) outB[idx] = (bf16)o;
  }
}

// ---------------------------------------------------------------------------
// Host side
// ---------------------------------------------------------------------------
extern "C" void kernel_launch(void* const* d_in, const int* in_sizes, int n_in,
                              void* d_out, int out_size, void* d_ws,
                              size_t ws_size, hipStream_t stream) {
  (void)in_sizes; (void)n_in; (void)out_size; (void)ws_size;
  const float* x    = (const float*)d_in[0];
  const float* mask = (const float*)d_in[1];
  const float* Wq = (const float*)d_in[2];  const float* bq = (const float*)d_in[3];
  const float* Wk = (const float*)d_in[4];  const float* bk = (const float*)d_in[5];
  const float* Wv = (const float*)d_in[6];  const float* bv = (const float*)d_in[7];
  const float* Wo = (const float*)d_in[8];  const float* bo = (const float*)d_in[9];
  const float* W1 = (const float*)d_in[10]; const float* b1 = (const float*)d_in[11];
  const float* W2 = (const float*)d_in[12]; const float* b2 = (const float*)d_in[13];
  const float* g1 = (const float*)d_in[14]; const float* be1 = (const float*)d_in[15];
  const float* g2 = (const float*)d_in[16]; const float* be2 = (const float*)d_in[17];
  float* out = (float*)d_out;

  char* ws = (char*)d_ws;
  const size_t szH  = (size_t)BS * HDIM;   // 8.39M elems
  const size_t szW  = (size_t)HDIM * HDIM;
  const size_t szW1 = (size_t)HDIM * FFDIM;
  size_t o = 0;
  bf16* xb  = (bf16*)(ws + o); o += szH * 2;
  bf16* WqT = (bf16*)(ws + o); o += szW * 2;
  bf16* WkT = (bf16*)(ws + o); o += szW * 2;
  bf16* WvT = (bf16*)(ws + o); o += szW * 2;
  bf16* WoT = (bf16*)(ws + o); o += szW * 2;
  bf16* W1T = (bf16*)(ws + o); o += szW1 * 2;
  bf16* W2T = (bf16*)(ws + o); o += szW1 * 2;
  bf16* qb  = (bf16*)(ws + o); o += szH * 2;
  bf16* kbp = (bf16*)(ws + o); o += szH * 2;
  bf16* vbp = (bf16*)(ws + o); o += szH * 2;
  bf16* cbp = (bf16*)(ws + o); o += szH * 2;
  bf16* gf  = qb;  // alias: q/k/v/ctx region (64MB+) reused for FF1 output
  float* t0 = (float*)(ws + o); o += szH * 4;
  float* hbuf = (float*)(ws + o); o += szH * 4;
  bf16* hb  = (bf16*)(ws + o); o += szH * 2;   // total ~184 MiB

  // 1. converts
  k_to_bf16<<<(int)(szH / 256), 256, 0, stream>>>(x, xb, szH);
  {
    dim3 g(HDIM / 32, HDIM / 32);
    k_transpose_bf16<<<g, 256, 0, stream>>>(Wq, WqT, HDIM, HDIM);
    k_transpose_bf16<<<g, 256, 0, stream>>>(Wk, WkT, HDIM, HDIM);
    k_transpose_bf16<<<g, 256, 0, stream>>>(Wv, WvT, HDIM, HDIM);
    k_transpose_bf16<<<g, 256, 0, stream>>>(Wo, WoT, HDIM, HDIM);
    dim3 g1d(FFDIM / 32, HDIM / 32);   // W1 [H,FF] -> [FF,H]
    k_transpose_bf16<<<g1d, 256, 0, stream>>>(W1, W1T, HDIM, FFDIM);
    dim3 g2d(HDIM / 32, FFDIM / 32);   // W2 [FF,H] -> [H,FF]
    k_transpose_bf16<<<g2d, 256, 0, stream>>>(W2, W2T, FFDIM, HDIM);
  }
  // 2. QKV projections (bf16 out)
  {
    dim3 g(HDIM / 128, BS / 128);
    k_gemm_bf16<false><<<g, 256, 0, stream>>>(xb, WqT, bq, nullptr, qb,  BS, HDIM, HDIM);
    k_gemm_bf16<false><<<g, 256, 0, stream>>>(xb, WkT, bk, nullptr, kbp, BS, HDIM, HDIM);
    k_gemm_bf16<false><<<g, 256, 0, stream>>>(xb, WvT, bv, nullptr, vbp, BS, HDIM, HDIM);
  }
  // 3. attention
  k_attention<<<BB * NHEAD * (SS / 128), 256, 0, stream>>>(qb, kbp, vbp, mask, cbp);
  // 4. output projection (f32 out)
  {
    dim3 g(HDIM / 128, BS / 128);
    k_gemm_bf16<false><<<g, 256, 0, stream>>>(cbp, WoT, bo, t0, nullptr, BS, HDIM, HDIM);
  }
  // 5. LN1(x + attn) -> h (f32) + hb (bf16)
  k_ln_residual<<<BS, 256, 0, stream>>>(x, t0, g1, be1, hbuf, hb);
  // 6. FF1 + GELU (bf16 out, aliases q/k/v/ctx region)
  {
    dim3 g(FFDIM / 128, BS / 128);
    k_gemm_bf16<true><<<g, 256, 0, stream>>>(hb, W1T, b1, nullptr, gf, BS, FFDIM, HDIM);
  }
  // 7. FF2 (f32 out)
  {
    dim3 g(HDIM / 128, BS / 128);
    k_gemm_bf16<false><<<g, 256, 0, stream>>>(gf, W2T, b2, t0, nullptr, BS, HDIM, FFDIM);
  }
  // 8. LN2(h + ff) -> out
  k_ln_residual<<<BS, 256, 0, stream>>>(hbuf, t0, g2, be2, out, nullptr);
}
